// WavKANEncoder_48180943126783
// MI455X (gfx1250) — matrix operand order repaired and checked by reading, not compiled, not run on hardware
//
#include <hip/hip_runtime.h>
#include <hip/hip_bf16.h>
#include <math.h>

typedef __bf16 bf16_t;
typedef __attribute__((ext_vector_type(16))) __bf16 v16bf;
typedef __attribute__((ext_vector_type(8)))  __bf16 v8bf;
typedef __attribute__((ext_vector_type(8)))  float  v8f;

__device__ __forceinline__ float gelu_exact(float x) {
    return 0.5f * x * (1.0f + erff(x * 0.70710678118654752f));
}

// ---------------------------------------------------------------------------
// prep: fold conv bias into BN affine; swizzle conv2/conv3 weights to bf16 in
// the per-lane A-fragment k order this code uses (contiguous k, B loads match):
//   A 16x32 bf16 layout: lanes 0-15 (row M=lane)   hold K {0..15}
//                        lanes 16-31 (row M=l-16)  hold K {16..31}
//   w2A[kk][mt(4)][lane(32)][j(16)]      (K = ci)
//   w3A[kk][kt(2)][mt(8)][lane(32)][j(16)] (K = ci - kt*32)
// ---------------------------------------------------------------------------
__global__ void k_prep(const float* __restrict__ w2, const float* __restrict__ w3,
                       const float* c2b, const float* g2, const float* b2,
                       const float* m2, const float* v2,
                       const float* c3b, const float* g3, const float* b3,
                       const float* m3, const float* v3,
                       bf16_t* __restrict__ w2A, bf16_t* __restrict__ w3A,
                       float* scale2, float* shift2, float* scale3, float* shift3) {
    int tid = blockIdx.x * blockDim.x + threadIdx.x;
    int nt  = gridDim.x * blockDim.x;
    for (int i = tid; i < 5 * 4 * 32 * 16; i += nt) {      // 10240
        int j    = i & 15;
        int lane = (i >> 4) & 31;
        int mt   = (i >> 9) & 3;
        int kk   = i >> 11;
        int co   = mt * 16 + (lane & 15);
        int hi   = lane >> 4;
        int base = hi * 16;
        int ci   = base + j;
        w2A[i] = (bf16_t)w2[(co * 32 + ci) * 5 + kk];
    }
    for (int i = tid; i < 3 * 2 * 8 * 32 * 16; i += nt) {  // 24576
        int j    = i & 15;
        int lane = (i >> 4) & 31;
        int mt   = (i >> 9) & 7;
        int kt   = (i >> 12) & 1;
        int kk   = i >> 13;
        int co   = mt * 16 + (lane & 15);
        int hi   = lane >> 4;
        int base = kt * 32 + hi * 16;
        int ci   = base + j;
        w3A[i] = (bf16_t)w3[(co * 64 + ci) * 3 + kk];
    }
    for (int i = tid; i < 64; i += nt) {
        float s = g2[i] * rsqrtf(v2[i] + 1e-5f);
        scale2[i] = s; shift2[i] = (c2b[i] - m2[i]) * s + b2[i];
    }
    for (int i = tid; i < 128; i += nt) {
        float s = g3[i] * rsqrtf(v3[i] + 1e-5f);
        scale3[i] = s; shift3[i] = (c3b[i] - m3[i]) * s + b3[i];
    }
}

// ---------------------------------------------------------------------------
// conv1 (Cin=1,K=7,pad=3) + BN + GELU + maxpool2  ->  h1 bf16 [B][32][512]
// ---------------------------------------------------------------------------
__global__ void __launch_bounds__(256) k_conv1(
        const float* __restrict__ x, const float* __restrict__ w,
        const float* __restrict__ cb, const float* __restrict__ g,
        const float* __restrict__ bb, const float* __restrict__ m,
        const float* __restrict__ v, bf16_t* __restrict__ h1) {
    int t  = blockIdx.x * 256 + threadIdx.x;     // B*32*512 threads
    int lp = t & 511;
    int co = (t >> 9) & 31;
    int b  = t >> 14;
    const float* xb = x + b * 1024;
    float wk[7];
#pragma unroll
    for (int k = 0; k < 7; k++) wk[k] = w[co * 7 + k];
    float sc = g[co] * rsqrtf(v[co] + 1e-5f);
    float sh = (cb[co] - m[co]) * sc + bb[co];
    float a0 = 0.f, a1 = 0.f;
    int l0 = lp * 2;
#pragma unroll
    for (int k = 0; k < 7; k++) {
        int p0 = l0 + k - 3;
        int p1 = p0 + 1;
        float x0 = (p0 >= 0 && p0 < 1024) ? xb[p0] : 0.f;
        float x1 = (p1 >= 0 && p1 < 1024) ? xb[p1] : 0.f;
        a0 += wk[k] * x0;
        a1 += wk[k] * x1;
    }
    float r0 = gelu_exact(a0 * sc + sh);
    float r1 = gelu_exact(a1 * sc + sh);
    h1[t] = (bf16_t)fmaxf(r0, r1);
}

// ---------------------------------------------------------------------------
// conv2 (32->64,K=5,pad=2) implicit GEMM via WMMA bf16 + BN + GELU + maxpool2
// h1 [b][ci][512] -> h2 [b][pos(256)][ci(64)]
// LDS staged as [sp(36)][ci(32)] so B fragments are contiguous 32B loads.
// ---------------------------------------------------------------------------
__global__ void __launch_bounds__(256) k_conv2(
        const bf16_t* __restrict__ h1, const bf16_t* __restrict__ w2A,
        const float* __restrict__ scale2, const float* __restrict__ shift2,
        bf16_t* __restrict__ h2) {
    __shared__ __align__(32) bf16_t sh1[36 * 32];   // [sp][ci]
    int b  = blockIdx.x >> 4;
    int nb = (blockIdx.x & 15) * 32;                // base position of block
    const bf16_t* h1b = h1 + b * 32 * 512;
    for (int i = threadIdx.x; i < 36 * 32; i += 256) {
        int ci = i / 36, sp = i % 36;               // global-coalesced fill
        int pos = nb - 2 + sp;
        sh1[sp * 32 + ci] =
            (pos >= 0 && pos < 512) ? h1b[ci * 512 + pos] : (bf16_t)0.0f;
    }
    __syncthreads();

    int wv   = threadIdx.x >> 5;
    int lane = threadIdx.x & 31;
    int mt   = wv & 3;                              // Cout tile
    int nbw  = nb + ((wv >> 2) << 4);               // wave position base
    int lh   = lane & 15;
    int hi   = lane >> 4;
    int pos  = nbw + lh;

    v8f c = {0.f, 0.f, 0.f, 0.f, 0.f, 0.f, 0.f, 0.f};
#pragma unroll
    for (int kk = 0; kk < 5; kk++) {
        v16bf a  = *(const v16bf*)(w2A + ((kk * 4 + mt) * 32 + lane) * 16);
        int   sp = pos + kk - nb;                   // 0..35
        v16bf bm = *(const v16bf*)(&sh1[sp * 32 + hi * 16]);
        c = __builtin_amdgcn_wmma_f32_16x16x32_bf16(false, a, false, bm,
                                                    (short)0, c, false, false);
    }

    // BN + GELU + maxpool2; pack 8 consecutive channels -> one 16B store
    bf16_t* h2b = h2 + b * 256 * 64;
    v8bf pk;
#pragma unroll
    for (int r = 0; r < 8; r++) {
        int co    = mt * 16 + hi * 8 + r;
        float val = gelu_exact(c[r] * scale2[co] + shift2[co]);
        float oth = __shfl_xor(val, 1);
        pk[r] = (bf16_t)fmaxf(val, oth);
    }
    if ((lane & 1) == 0)
        *(v8bf*)(&h2b[(pos >> 1) * 64 + mt * 16 + hi * 8]) = pk;
}

// ---------------------------------------------------------------------------
// conv3 (64->128,K=3,pad=1) WMMA + BN + GELU + avgpool(16)  ->  feat f32
// h2 [b][pos][ci] ; LDS staged as [sp(18)][ci(64)] (contiguous fill + loads)
// ---------------------------------------------------------------------------
__global__ void __launch_bounds__(256) k_conv3(
        const bf16_t* __restrict__ h2, const bf16_t* __restrict__ w3A,
        const float* __restrict__ scale3, const float* __restrict__ shift3,
        float* __restrict__ feat) {
    __shared__ __align__(32) bf16_t sh2[18 * 64];   // [sp][ci]
    int b   = blockIdx.x >> 4;
    int bin = blockIdx.x & 15;
    int nb  = bin * 16;
    const bf16_t* h2b = h2 + b * 256 * 64;
    for (int i = threadIdx.x; i < 18 * 64; i += 256) {
        int sp = i / 64, ci = i % 64;
        int pos = nb - 1 + sp;
        sh2[i] = (pos >= 0 && pos < 256) ? h2b[pos * 64 + ci] : (bf16_t)0.0f;
    }
    __syncthreads();

    int wv   = threadIdx.x >> 5;                    // Cout tile 0..7
    int lane = threadIdx.x & 31;
    int lh   = lane & 15;
    int hi   = lane >> 4;

    v8f c = {0.f, 0.f, 0.f, 0.f, 0.f, 0.f, 0.f, 0.f};
#pragma unroll
    for (int kk = 0; kk < 3; kk++) {
#pragma unroll
        for (int kt = 0; kt < 2; kt++) {
            v16bf a = *(const v16bf*)(
                w3A + (((kk * 2 + kt) * 8 + wv) * 32 + lane) * 16);
            int   sp = lh + kk;                     // 0..17
            v16bf bm = *(const v16bf*)(&sh2[sp * 64 + kt * 32 + hi * 16]);
            c = __builtin_amdgcn_wmma_f32_16x16x32_bf16(false, a, false, bm,
                                                        (short)0, c, false, false);
        }
    }
#pragma unroll
    for (int r = 0; r < 8; r++) {
        int co    = wv * 16 + hi * 8 + r;
        float val = gelu_exact(c[r] * scale3[co] + shift3[co]);
        val += __shfl_xor(val, 1);
        val += __shfl_xor(val, 2);
        val += __shfl_xor(val, 4);
        val += __shfl_xor(val, 8);                  // sum over 16 positions
        if (lh == 0) feat[b * 2048 + co * 16 + bin] = val * (1.f / 16.f);
    }
}

// ---------------------------------------------------------------------------
// fused wavelet-KAN x3 + SiLU + LayerNorm; block = one batch row
// ---------------------------------------------------------------------------
__device__ __forceinline__ void silu_ln64(float* in64, float* out64,
                                          const float* g, const float* bb, int tid) {
    if (tid < 64) {
        float v = in64[tid];
        in64[tid] = v / (1.f + expf(-v));
    }
    __syncthreads();
    float m = 0.f;
    for (int i = 0; i < 64; i++) m += in64[i];
    m *= (1.f / 64.f);
    float var = 0.f;
    for (int i = 0; i < 64; i++) { float d = in64[i] - m; var += d * d; }
    var *= (1.f / 64.f);
    if (tid < 64)
        out64[tid] = (in64[tid] - m) * rsqrtf(var + 1e-5f) * g[tid] + bb[tid];
    __syncthreads();
}

__global__ void __launch_bounds__(256) k_wav(
        const float* __restrict__ feat,
        const float* t1, const float* s1, const float* w1, const float* g1, const float* b1,
        const float* t2, const float* s2, const float* w2, const float* g2, const float* b2,
        const float* t3, const float* s3, const float* w3, const float* g3, const float* b3,
        float* __restrict__ out) {
    __shared__ float sf[2048];
    __shared__ float sa[64];
    __shared__ float sb[64];
    int b    = blockIdx.x;
    int tid  = threadIdx.x;
    int wv   = tid >> 5;
    int lane = tid & 31;
    for (int i = tid; i < 2048; i += 256) sf[i] = feat[b * 2048 + i];
    __syncthreads();

    // layer 1: in=2048, out=64 (wave per output)
#pragma unroll 1
    for (int q = 0; q < 8; q++) {
        int o = wv * 8 + q;
        const float* tr = t1 + o * 2048;
        const float* sr = s1 + o * 2048;
        const float* wr = w1 + o * 2048;
        float acc = 0.f;
        for (int i = lane; i < 2048; i += 32) {
            float z  = (sf[i] - tr[i]) / (sr[i] + 1e-8f);
            float z2 = z * z;
            acc += (1.f - z2) * expf(-0.5f * z2) * wr[i];
        }
        acc += __shfl_xor(acc, 16); acc += __shfl_xor(acc, 8);
        acc += __shfl_xor(acc, 4);  acc += __shfl_xor(acc, 2);
        acc += __shfl_xor(acc, 1);
        if (lane == 0) sa[o] = acc;
    }
    __syncthreads();
    silu_ln64(sa, sb, g1, b1, tid);

    // layer 2: in=64
#pragma unroll 1
    for (int q = 0; q < 8; q++) {
        int o = wv * 8 + q;
        float acc = 0.f;
        for (int i = lane; i < 64; i += 32) {
            float z  = (sb[i] - t2[o * 64 + i]) / (s2[o * 64 + i] + 1e-8f);
            float z2 = z * z;
            acc += (1.f - z2) * expf(-0.5f * z2) * w2[o * 64 + i];
        }
        acc += __shfl_xor(acc, 16); acc += __shfl_xor(acc, 8);
        acc += __shfl_xor(acc, 4);  acc += __shfl_xor(acc, 2);
        acc += __shfl_xor(acc, 1);
        if (lane == 0) sa[o] = acc;
    }
    __syncthreads();
    silu_ln64(sa, sb, g2, b2, tid);

    // layer 3: in=64
#pragma unroll 1
    for (int q = 0; q < 8; q++) {
        int o = wv * 8 + q;
        float acc = 0.f;
        for (int i = lane; i < 64; i += 32) {
            float z  = (sb[i] - t3[o * 64 + i]) / (s3[o * 64 + i] + 1e-8f);
            float z2 = z * z;
            acc += (1.f - z2) * expf(-0.5f * z2) * w3[o * 64 + i];
        }
        acc += __shfl_xor(acc, 16); acc += __shfl_xor(acc, 8);
        acc += __shfl_xor(acc, 4);  acc += __shfl_xor(acc, 2);
        acc += __shfl_xor(acc, 1);
        if (lane == 0) sa[o] = acc;
    }
    __syncthreads();
    silu_ln64(sa, sb, g3, b3, tid);
    if (tid < 64) out[b * 64 + tid] = sb[tid];
}

// ---------------------------------------------------------------------------
extern "C" void kernel_launch(void* const* d_in, const int* in_sizes, int n_in,
                              void* d_out, int out_size, void* d_ws, size_t ws_size,
                              hipStream_t stream) {
    const float* x       = (const float*)d_in[0];
    const float* conv1_w = (const float*)d_in[1];
    const float* conv1_b = (const float*)d_in[2];
    const float* bn1_g   = (const float*)d_in[3];
    const float* bn1_b   = (const float*)d_in[4];
    const float* bn1_m   = (const float*)d_in[5];
    const float* bn1_v   = (const float*)d_in[6];
    const float* conv2_w = (const float*)d_in[7];
    const float* conv2_b = (const float*)d_in[8];
    const float* bn2_g   = (const float*)d_in[9];
    const float* bn2_b   = (const float*)d_in[10];
    const float* bn2_m   = (const float*)d_in[11];
    const float* bn2_v   = (const float*)d_in[12];
    const float* conv3_w = (const float*)d_in[13];
    const float* conv3_b = (const float*)d_in[14];
    const float* bn3_g   = (const float*)d_in[15];
    const float* bn3_b   = (const float*)d_in[16];
    const float* bn3_m   = (const float*)d_in[17];
    const float* bn3_v   = (const float*)d_in[18];
    const float* wl1_t = (const float*)d_in[19];
    const float* wl1_s = (const float*)d_in[20];
    const float* wl1_w = (const float*)d_in[21];
    const float* ln1_g = (const float*)d_in[22];
    const float* ln1_b = (const float*)d_in[23];
    const float* wl2_t = (const float*)d_in[24];
    const float* wl2_s = (const float*)d_in[25];
    const float* wl2_w = (const float*)d_in[26];
    const float* ln2_g = (const float*)d_in[27];
    const float* ln2_b = (const float*)d_in[28];
    const float* wl3_t = (const float*)d_in[29];
    const float* wl3_s = (const float*)d_in[30];
    const float* wl3_w = (const float*)d_in[31];
    const float* ln3_g = (const float*)d_in[32];
    const float* ln3_b = (const float*)d_in[33];
    float* out = (float*)d_out;

    // workspace layout (256B-aligned offsets)
    char* ws = (char*)d_ws;
    bf16_t* w2A    = (bf16_t*)(ws);                       // 20480 B
    bf16_t* w3A    = (bf16_t*)(ws + 20480);               // 49152 B
    float*  scale2 = (float*)(ws + 69632);                // 256 B
    float*  shift2 = (float*)(ws + 69888);                // 256 B
    float*  scale3 = (float*)(ws + 70144);                // 512 B
    float*  shift3 = (float*)(ws + 70656);                // 512 B
    bf16_t* h1     = (bf16_t*)(ws + 71168);               // 32 MB
    bf16_t* h2     = (bf16_t*)(ws + 71168 + 33554432);    // 32 MB
    float*  feat   = (float*)(ws + 71168);                // aliases h1 (dead by then)

    k_prep<<<96, 256, 0, stream>>>(conv2_w, conv3_w,
                                   conv2_b, bn2_g, bn2_b, bn2_m, bn2_v,
                                   conv3_b, bn3_g, bn3_b, bn3_m, bn3_v,
                                   w2A, w3A, scale2, shift2, scale3, shift3);

    k_conv1<<<65536, 256, 0, stream>>>(x, conv1_w, conv1_b,
                                       bn1_g, bn1_b, bn1_m, bn1_v, h1);

    k_conv2<<<1024 * 16, 256, 0, stream>>>(h1, w2A, scale2, shift2, h2);

    k_conv3<<<1024 * 16, 256, 0, stream>>>(h2, w3A, scale3, shift3, feat);

    k_wav<<<1024, 256, 0, stream>>>(feat,
                                    wl1_t, wl1_s, wl1_w, ln1_g, ln1_b,
                                    wl2_t, wl2_s, wl2_w, ln2_g, ln2_b,
                                    wl3_t, wl3_s, wl3_w, ln3_g, ln3_b,
                                    out);
}
